// MixtralAttention_27573690040859
// MI455X (gfx1250) — matrix operand order repaired and checked by reading, not compiled, not run on hardware
//
#include <hip/hip_runtime.h>

// ---------------------------------------------------------------------------
// MixtralAttention for gfx1250 (CDNA5, wave32, WMMA).
// Pipeline: GEMM(qkv) -> RoPE -> flash attention (GQA-shared K/V) -> GEMM(out).
// All big matmuls use v_wmma_f32_16x16x32_bf16 (fp32 accumulate).
// ---------------------------------------------------------------------------

typedef __attribute__((ext_vector_type(8)))  float  v8f;
typedef __attribute__((ext_vector_type(16))) __bf16 v16bf;
typedef __attribute__((ext_vector_type(8)))  __bf16 v8bf;
typedef __attribute__((ext_vector_type(4)))  __bf16 v4bf;

constexpr int B_   = 2;
constexpr int S_   = 2048;
constexpr int H_   = 4096;
constexpr int NH_  = 32;
constexpr int NKV_ = 8;
constexpr int HD_  = 128;
constexpr int QKVN = NH_ * HD_ + 2 * NKV_ * HD_;   // 6144
constexpr int BS   = B_ * S_;                      // 4096
constexpr float SCALE_ = 0.08838834764831845f;     // 1/sqrt(128)
// sliding window (4096) >= S (2048) => pure causal mask.

// ---- helpers --------------------------------------------------------------

__device__ __forceinline__ float ldA_(const float* p, size_t i) { return p[i]; }
__device__ __forceinline__ float ldA_(const unsigned short* p, size_t i) {
  __bf16 v; __builtin_memcpy(&v, p + i, 2); return (float)v;
}

// Build a 16-element bf16 fragment from two contiguous 8-element LDS runs.
// hiOff = element offset of the second run (16 for both A- and B-layout).
__device__ __forceinline__ v16bf frag_(const __bf16* p, int hiOff) {
  v8bf lo = *(const v8bf*)p;
  v8bf hi = *(const v8bf*)(p + hiOff);
  return __builtin_shufflevector(lo, hi, 0,1,2,3,4,5,6,7,8,9,10,11,12,13,14,15);
}

// ---- GEMM: C[M,N] = A[M,K] * B[K,N], bf16 WMMA, fp32 accumulate -----------
// 128x128 block tile, 8 waves (4x2), each wave 32x64 (2x4 WMMA tiles), BK=32.
// Register-buffered pipeline: global loads for tile k+1 overlap WMMAs of k.

template <typename TA>
__global__ __launch_bounds__(256)
void gemm_bf16_wmma(const TA* __restrict__ A, const float* __restrict__ Bm,
                    float* __restrict__ C, int M, int N, int K) {
  __shared__ __align__(16) __bf16 As[128 * 40];   // row-major, pad 32->40
  __shared__ __align__(16) __bf16 Bs[128 * 40];   // column-major, pad 32->40

  const int tid  = threadIdx.x;
  const int wave = tid >> 5;
  const int lane = tid & 31;
  const int half = lane >> 4;
  const int lr   = lane & 15;
  const int wm   = wave >> 1;     // wave row (0..3)
  const int wn   = wave & 1;      // wave col (0..1)
  const int m0   = blockIdx.y * 128;
  const int n0   = blockIdx.x * 128;

  const int ar = tid >> 3;        // A stage: base row 0..31 (4 rows apart 32)
  const int ac = (tid & 7) * 4;   // A stage: 4 contiguous cols

  float aReg[16], bReg[16];

  auto loadTiles = [&](int k0) {
    #pragma unroll
    for (int i = 0; i < 4; ++i) {
      size_t base = (size_t)(m0 + ar + i * 32) * K + k0 + ac;
      #pragma unroll
      for (int jj = 0; jj < 4; ++jj) aReg[i * 4 + jj] = ldA_(A, base + jj);
    }
    #pragma unroll
    for (int i = 0; i < 4; ++i) {
      int g = tid + i * 256;
      int r = g >> 5;              // 0..31
      int c = (g & 31) * 4;        // 0..124
      const float4 f = *(const float4*)&Bm[(size_t)(k0 + r) * N + n0 + c];
      bReg[i * 4 + 0] = f.x; bReg[i * 4 + 1] = f.y;
      bReg[i * 4 + 2] = f.z; bReg[i * 4 + 3] = f.w;
    }
  };
  auto storeTiles = [&]() {
    #pragma unroll
    for (int i = 0; i < 4; ++i) {
      v4bf o;
      #pragma unroll
      for (int jj = 0; jj < 4; ++jj) o[jj] = (__bf16)aReg[i * 4 + jj];
      *(v4bf*)&As[(ar + i * 32) * 40 + ac] = o;
    }
    #pragma unroll
    for (int i = 0; i < 4; ++i) {
      int g = tid + i * 256;
      int r = g >> 5;
      int c = (g & 31) * 4;
      #pragma unroll
      for (int jj = 0; jj < 4; ++jj)
        Bs[(c + jj) * 40 + r] = (__bf16)bReg[i * 4 + jj];
    }
  };

  v8f acc[2][4];
  #pragma unroll
  for (int t = 0; t < 2; ++t)
    #pragma unroll
    for (int u = 0; u < 4; ++u)
      #pragma unroll
      for (int r = 0; r < 8; ++r) acc[t][u][r] = 0.0f;

  loadTiles(0);
  for (int k0 = 0; k0 < K; k0 += 32) {
    storeTiles();
    __syncthreads();
    if (k0 + 32 < K) loadTiles(k0 + 32);   // overlaps with compute below

    v16bf a[2], b[4];
    // A fragment: row = lr, elems 0..7 -> k=half*8+0..7, 8..15 -> +16.
    #pragma unroll
    for (int t = 0; t < 2; ++t)
      a[t] = frag_(&As[(wm * 32 + t * 16 + lr) * 40 + half * 8], 16);
    // B fragment: col = lr, elems 0..7 -> k=half*8+0..7, 8..15 -> +16.
    #pragma unroll
    for (int u = 0; u < 4; ++u)
      b[u] = frag_(&Bs[(wn * 64 + u * 16 + lr) * 40 + half * 8], 16);

    #pragma unroll
    for (int t = 0; t < 2; ++t)
      #pragma unroll
      for (int u = 0; u < 4; ++u)
        acc[t][u] = __builtin_amdgcn_wmma_f32_16x16x32_bf16(
            false, a[t], false, b[u], (short)0, acc[t][u], false, false);
    __syncthreads();
  }

  // C layout: VGPR r holds M = r + 8*half, column = lr.
  #pragma unroll
  for (int t = 0; t < 2; ++t)
    #pragma unroll
    for (int u = 0; u < 4; ++u) {
      int row = m0 + wm * 32 + t * 16 + half * 8;
      int col = n0 + wn * 64 + u * 16 + lr;
      #pragma unroll
      for (int r = 0; r < 8; ++r)
        C[(size_t)(row + r) * N + col] = acc[t][u][r];
    }
}

// ---- RoPE (interleaved / GPT-J style) on q and k slices of qkv ------------

__global__ void rope_kernel(float* __restrict__ qkv, const int* __restrict__ pos) {
  long tid = (long)blockIdx.x * blockDim.x + threadIdx.x;
  const long total = (long)B_ * S_ * (NH_ + NKV_) * (HD_ / 2);
  if (tid >= total) return;
  int d       = (int)(tid & 63);
  long rest   = tid >> 6;
  int hIdx    = (int)(rest % (NH_ + NKV_));
  long sr     = rest / (NH_ + NKV_);
  int s       = (int)(sr % S_);
  int b       = (int)(sr / S_);
  float p     = (float)pos[b * S_ + s];
  // 10000^(-2d/128) = exp(-d * ln(10000)/64)
  float inv   = __expf((float)d * -0.14391156831212787f);
  float ang   = p * inv;
  float sn, cs;
  __sincosf(ang, &sn, &cs);
  size_t base = ((size_t)b * S_ + s) * QKVN +
                (hIdx < NH_ ? hIdx * HD_ : NH_ * HD_ + (hIdx - NH_) * HD_) + 2 * d;
  float x1 = qkv[base], x2 = qkv[base + 1];
  qkv[base]     = x1 * cs - x2 * sn;
  qkv[base + 1] = x2 * cs + x1 * sn;
}

// ---- Flash attention ------------------------------------------------------
// Block = (batch, kv-head, 16-query tile); the 4 waves are the 4 query heads
// of the GQA group, sharing one K/V tile staged in LDS (f32 -> bf16 fused).
// Chunk-loop trip count is block-uniform, so __syncthreads inside is legal.

__global__ __launch_bounds__(128)
void attn_kernel(const float* __restrict__ qkv, unsigned short* __restrict__ attn) {
  __shared__ __align__(16) __bf16 Ks[32 * 136];      // K tile, row-major pad 136
  __shared__ __align__(16) __bf16 Vs[128 * 40];      // V tile, transposed pad 40
  __shared__ __align__(16) __bf16 Plds[4 * 16 * 40]; // per-wave P staging

  const int tid  = threadIdx.x;
  const int wave = tid >> 5;
  const int lane = tid & 31;
  const int half = lane >> 4;
  const int lr   = lane & 15;

  const int qt   = blockIdx.x & 127;   // S/16 = 128 query tiles
  const int bkv  = blockIdx.x >> 7;
  const int b    = bkv >> 3;           // NKV = 8
  const int kvh  = bkv & 7;
  const int h    = kvh * 4 + wave;     // this wave's query head
  const int qbase = qt * 16;

  const float* qp = qkv + (size_t)b * S_ * QKVN + h * HD_;
  const float* kp = qkv + (size_t)b * S_ * QKVN + NH_ * HD_ + kvh * HD_;
  const float* vp = qkv + (size_t)b * S_ * QKVN + NH_ * HD_ + NKV_ * HD_ + kvh * HD_;
  __bf16* pl = &Plds[wave * 16 * 40];

  // Q fragments (A layout): 4 chunks of hd=32, row = lr of the q tile.
  v16bf qf[4];
  {
    const float* qrow = qp + (size_t)(qbase + lr) * QKVN;
    #pragma unroll
    for (int kk = 0; kk < 4; ++kk) {
      int c0 = kk * 32 + half * 8;
      const float4* f0 = (const float4*)(qrow + c0);
      const float4* f1 = (const float4*)(qrow + c0 + 16);
      float4 x0 = f0[0], x1 = f0[1], y0 = f1[0], y1 = f1[1];
      v16bf a;
      a[0]=(__bf16)x0.x;  a[1]=(__bf16)x0.y;  a[2]=(__bf16)x0.z;  a[3]=(__bf16)x0.w;
      a[4]=(__bf16)x1.x;  a[5]=(__bf16)x1.y;  a[6]=(__bf16)x1.z;  a[7]=(__bf16)x1.w;
      a[8]=(__bf16)y0.x;  a[9]=(__bf16)y0.y;  a[10]=(__bf16)y0.z; a[11]=(__bf16)y0.w;
      a[12]=(__bf16)y1.x; a[13]=(__bf16)y1.y; a[14]=(__bf16)y1.z; a[15]=(__bf16)y1.w;
      qf[kk] = a;
    }
  }

  v8f oacc[8];
  #pragma unroll
  for (int c = 0; c < 8; ++c)
    #pragma unroll
    for (int r = 0; r < 8; ++r) oacc[c][r] = 0.0f;
  float mrow[8], lrow[8];
  #pragma unroll
  for (int r = 0; r < 8; ++r) { mrow[r] = -3.0e38f; lrow[r] = 0.0f; }

  const int nchunks = (qt + 2) >> 1;   // ceil((qbase+16)/32); keys stay < S
  for (int j = 0; j < nchunks; ++j) {
    const int kbase = j * 32;

    // ---- stage K (row-major) and V (transposed) tiles into LDS as bf16 --
    #pragma unroll
    for (int i = 0; i < 8; ++i) {
      int g = tid + i * 128;
      int r = g >> 5;              // key row 0..31
      int c = (g & 31) * 4;        // hd col group
      float4 f = *(const float4*)(kp + (size_t)(kbase + r) * QKVN + c);
      v4bf o;
      o[0]=(__bf16)f.x; o[1]=(__bf16)f.y; o[2]=(__bf16)f.z; o[3]=(__bf16)f.w;
      *(v4bf*)&Ks[r * 136 + c] = o;
    }
    #pragma unroll
    for (int i = 0; i < 8; ++i) {
      int g = tid + i * 128;
      int r = g >> 5;
      int c = (g & 31) * 4;
      float4 f = *(const float4*)(vp + (size_t)(kbase + r) * QKVN + c);
      Vs[(c + 0) * 40 + r] = (__bf16)f.x;
      Vs[(c + 1) * 40 + r] = (__bf16)f.y;
      Vs[(c + 2) * 40 + r] = (__bf16)f.z;
      Vs[(c + 3) * 40 + r] = (__bf16)f.w;
    }
    __syncthreads();
    if (j + 1 < nchunks)
      __builtin_prefetch(kp + (size_t)(kbase + 32 + (tid >> 2)) * QKVN, 0, 1);

    // ---- scores: two 16-key tiles, each = sum of 4 WMMAs over hd --------
    v8f sc[2];
    #pragma unroll
    for (int tkb = 0; tkb < 2; ++tkb) {
      const int key = kbase + tkb * 16 + lr;       // B-frag column
      v8f s;
      #pragma unroll
      for (int r = 0; r < 8; ++r) s[r] = 0.0f;
      #pragma unroll
      for (int kk = 0; kk < 4; ++kk) {
        v16bf bfr = frag_(&Ks[(tkb * 16 + lr) * 136 + kk * 32 + half * 8], 16);
        s = __builtin_amdgcn_wmma_f32_16x16x32_bf16(
            false, qf[kk], false, bfr, (short)0, s, false, false);
      }
      // scale + causal mask (window >= S => pure causal)
      #pragma unroll
      for (int r = 0; r < 8; ++r) {
        int qi = qbase + r + 8 * half;
        s[r] = (key <= qi) ? s[r] * SCALE_ : -1.0e30f;
      }
      sc[tkb] = s;
    }

    // ---- online softmax (row = r + 8*half; 16 cols live in one half-wave)
    float alpha[8];
    #pragma unroll
    for (int r = 0; r < 8; ++r) {
      float rm = fmaxf(sc[0][r], sc[1][r]);
      #pragma unroll
      for (int msk = 1; msk < 16; msk <<= 1)
        rm = fmaxf(rm, __shfl_xor(rm, msk, 32));
      float mn = fmaxf(mrow[r], rm);
      float p0 = __expf(sc[0][r] - mn);
      float p1 = __expf(sc[1][r] - mn);
      float rs = p0 + p1;
      #pragma unroll
      for (int msk = 1; msk < 16; msk <<= 1)
        rs += __shfl_xor(rs, msk, 32);
      float al = __expf(mrow[r] - mn);
      lrow[r]  = lrow[r] * al + rs;
      mrow[r]  = mn;
      alpha[r] = al;
      int m = r + 8 * half;
      pl[m * 40 + lr]      = (__bf16)p0;   // P tile 16x32, row-major, pad 40
      pl[m * 40 + 16 + lr] = (__bf16)p1;
    }
    #pragma unroll
    for (int c = 0; c < 8; ++c)
      #pragma unroll
      for (int r = 0; r < 8; ++r) oacc[c][r] *= alpha[r];

    // C-layout -> A-layout relayout via LDS (same-wave RAW on DScnt)
    asm volatile("s_wait_dscnt 0" ::: "memory");
    v16bf pf = frag_(&pl[lr * 40 + half * 8], 16);

    // ---- P * V : 8 output hd chunks of 16 (V fragments from LDS) --------
    #pragma unroll
    for (int c = 0; c < 8; ++c) {
      v16bf vf = frag_(&Vs[(c * 16 + lr) * 40 + half * 8], 16);
      oacc[c] = __builtin_amdgcn_wmma_f32_16x16x32_bf16(
          false, pf, false, vf, (short)0, oacc[c], false, false);
    }
    __syncthreads();   // before next chunk overwrites Ks/Vs
  }

  // ---- epilogue: normalize and store bf16 for the output projection -----
  #pragma unroll
  for (int r = 0; r < 8; ++r) lrow[r] = 1.0f / lrow[r];
  #pragma unroll
  for (int c = 0; c < 8; ++c)
    #pragma unroll
    for (int r = 0; r < 8; ++r) {
      int m = r + 8 * half;
      __bf16 bv = (__bf16)(oacc[c][r] * lrow[r]);
      unsigned short u; __builtin_memcpy(&u, &bv, 2);
      attn[((size_t)b * S_ + qbase + m) * (size_t)H_ + h * HD_ + c * 16 + lr] = u;
    }
}

// ---------------------------------------------------------------------------

extern "C" void kernel_launch(void* const* d_in, const int* in_sizes, int n_in,
                              void* d_out, int out_size, void* d_ws, size_t ws_size,
                              hipStream_t stream) {
  const float* hs   = (const float*)d_in[0];   // [B,S,H] f32
  const int*   pos  = (const int*)d_in[1];     // [B,S] i32
  const float* wqkv = (const float*)d_in[2];   // [H, 6144] f32
  const float* wo   = (const float*)d_in[3];   // [4096, H] f32
  float* out = (float*)d_out;                  // [B,S,H] f32

  float* qkv = (float*)d_ws;                                              // 96 MB
  unsigned short* attn =
      (unsigned short*)((char*)d_ws + (size_t)BS * QKVN * sizeof(float)); // 32 MB

  // 1) QKV projection: qkv[BS,6144] = hs[BS,4096] x wqkv[4096,6144]
  dim3 g1(QKVN / 128, BS / 128);
  gemm_bf16_wmma<float><<<g1, 256, 0, stream>>>(hs, wqkv, qkv, BS, QKVN, H_);

  // 2) RoPE (interleaved) in-place on q and k slices
  long ropeTot = (long)B_ * S_ * (NH_ + NKV_) * (HD_ / 2);
  rope_kernel<<<(int)((ropeTot + 255) / 256), 256, 0, stream>>>(qkv, pos);

  // 3) Flash attention (GQA, causal): block = (b, kvh, qtile), 4 waves = 4 q heads
  attn_kernel<<<B_ * NKV_ * (S_ / 16), 128, 0, stream>>>(qkv, attn);

  // 4) Output projection: out[BS,4096] = attn[BS,4096] x wo[4096,4096]
  dim3 g2(H_ / 128, BS / 128);
  gemm_bf16_wmma<unsigned short><<<g2, 256, 0, stream>>>(attn, wo, out, BS, H_, H_);
}
